// VGraphEncoder_63814624084747
// MI455X (gfx1250) — compile-verified, matrix operand order checked by reading, not run on hardware
//
#include <hip/hip_runtime.h>

typedef __bf16 bf16_t;
typedef __attribute__((ext_vector_type(16))) __bf16 v16bf;
typedef __attribute__((ext_vector_type(8)))  float  v8f;

#define THREADS 256

// ---------------- degree / normalization ----------------

__global__ void k_deg_init(float* __restrict__ deg, int n) {
  int i = blockIdx.x * blockDim.x + threadIdx.x;
  if (i < n) deg[i] = 1.0f;  // self-loop contributes 1 to every node
}

__global__ void k_deg_count(const int* __restrict__ dst, float* __restrict__ deg, int e) {
  int i = blockIdx.x * blockDim.x + threadIdx.x;
  if (i < e) atomicAdd(&deg[dst[i]], 1.0f);
}

__global__ void k_dinv(float* __restrict__ deg, int n) {
  int i = blockIdx.x * blockDim.x + threadIdx.x;
  if (i < n) deg[i] = rsqrtf(deg[i]);  // deg >= 1 always (self-loops)
}

// ---------------- layer-1 GEMM: [N,512] x [512,64] via bf16 WMMA ----------------
// One block = 16 output rows; 4 waves each own a 16-col tile of the 64 cols.
// LDS holds tiles in WMMA *fragment order* (lane-major, 32B/lane):
//   reads  : one contiguous v16bf per lane  -> 2 x ds_load_b128
//   writes : pair-packed bf16 (consecutive K -> consecutive even slots)
//            A: 1 x ds_store_b64/thread, B: 8 x ds_store_b32/thread
// Fragment slot for K-index k (16-bit A/B 16x32 layout):
//   half = (k>>3)&1, khat = k-8*half, slot = khat<8 ? khat : khat-8 (lane = m/n + 16*half)

__global__ __launch_bounds__(128) void k_gemm_512x64_wmma(
    const float* __restrict__ X, const float* __restrict__ W, float* __restrict__ out) {
  __shared__ v16bf sAf[32];       // A fragments, lane-major (1 KB)
  __shared__ v16bf sBf[4][32];    // B fragments per wave tile (4 KB)
  unsigned* const aW32 = (unsigned*)sAf;   // 8 u32 words per lane fragment
  unsigned* const bW32 = (unsigned*)sBf;

  const int tid  = threadIdx.x;
  const int wave = tid >> 5;      // wave32: 4 waves per block
  const int lane = tid & 31;
  const int half = lane >> 4;
  const int idx  = lane & 15;
  const size_t rowbase = (size_t)blockIdx.x * 16;

  // --- A staging geometry (fixed across K-steps): 4 consecutive k, one row ---
  const int pA = tid * 4;                    // 0,4,...,508
  const int mA = pA >> 5, cA = pA & 31;      // cA multiple of 4 -> one half, even slot
  const int hA = (cA >> 3) & 1;
  const int khA = cA - 8 * hA;
  const int slA = (khA < 8) ? khA : khA - 8; // multiple of 4
  const int aWord = (mA + 16 * hA) * 8 + (slA >> 1);  // even -> b64-aligned

  // --- B staging geometry: k-pair (kB,kB+1) x 8 columns ---
  const int kB  = (tid >> 3) * 2;            // 0,2,...,30
  const int n0B = (tid & 7) * 8;             // 0,8,...,56
  const int hB  = (kB >> 3) & 1;
  const int khB = kB - 8 * hB;
  const int slB = (khB < 8) ? khB : khB - 8; // even
  const int bWordBase = slB >> 1;

  v8f acc = {};
  for (int kb = 0; kb < 512; kb += 32) {
    // A tile: one float4 load, one packed b64 LDS store
    {
      const float4 v4 = *(const float4*)(X + (rowbase + mA) * 512 + kb + cA);
      union { bf16_t h[2]; unsigned u; } p0, p1;
      p0.h[0] = (bf16_t)v4.x; p0.h[1] = (bf16_t)v4.y;     // slots slA, slA+1
      p1.h[0] = (bf16_t)v4.z; p1.h[1] = (bf16_t)v4.w;     // slots slA+2, slA+3
      *(uint2*)(aW32 + aWord) = make_uint2(p0.u, p1.u);
    }
    // B tile: rows kB, kB+1 x cols n0B..n0B+7 (4 coalesced float4 loads)
    {
      const float4 r0a = *(const float4*)(W + (size_t)(kb + kB) * 64 + n0B);
      const float4 r0b = *(const float4*)(W + (size_t)(kb + kB) * 64 + n0B + 4);
      const float4 r1a = *(const float4*)(W + (size_t)(kb + kB + 1) * 64 + n0B);
      const float4 r1b = *(const float4*)(W + (size_t)(kb + kB + 1) * 64 + n0B + 4);
      const float r0[8] = {r0a.x, r0a.y, r0a.z, r0a.w, r0b.x, r0b.y, r0b.z, r0b.w};
      const float r1[8] = {r1a.x, r1a.y, r1a.z, r1a.w, r1b.x, r1b.y, r1b.z, r1b.w};
#pragma unroll
      for (int j = 0; j < 8; ++j) {
        const int n = n0B + j;
        union { bf16_t h[2]; unsigned u; } pk;
        pk.h[0] = (bf16_t)r0[j];                           // slot slB   (k = kB)
        pk.h[1] = (bf16_t)r1[j];                           // slot slB+1 (k = kB+1)
        bW32[((n >> 4) * 32 + (n & 15) + 16 * hB) * 8 + bWordBase] = pk.u;
      }
    }
    __syncthreads();

    const v16bf a = sAf[lane];                 // 2 x ds_load_b128
    const v16bf b = sBf[wave][lane];           // 2 x ds_load_b128
    acc = __builtin_amdgcn_wmma_f32_16x16x32_bf16(false, a, false, b,
                                                  (short)0, acc, false, false);
    __syncthreads();
  }
#pragma unroll
  for (int r = 0; r < 8; ++r) {
    const int m = r + half * 8;                // C/D layout: VGPR r -> row r + 8*half
    out[(rowbase + m) * 64 + wave * 16 + idx] = acc[r];
  }
}

// ---------------- small dense layers (64->8, 8->4, 4->2): VALU dot ----------------

__global__ void k_linear(const float* __restrict__ in, const float* __restrict__ W,
                         float* __restrict__ out, int n, int fin, int fout, int relu_in) {
  int t = blockIdx.x * blockDim.x + threadIdx.x;
  if (t >= n * fout) return;
  int i = t / fout, fo = t % fout;
  const float* row = in + (size_t)i * fin;
  float s = 0.0f;
  for (int k = 0; k < fin; ++k) {
    float v = row[k];
    if (relu_in) v = fmaxf(v, 0.0f);
    s += v * W[k * fout + fo];
  }
  out[t] = s;
}

// ---------------- aggregation: bias + self-loop init, then edge scatter ----------------

template <int F>
__global__ void k_agg_init(const float* __restrict__ hlin, const float* __restrict__ dinv,
                           const float* __restrict__ bias, float* __restrict__ out, int n) {
  int t = blockIdx.x * blockDim.x + threadIdx.x;
  if (t >= n * F) return;
  int i = t / F, c = t % F;
  float di = dinv[i];
  out[t] = bias[c] + hlin[t] * di * di;   // self-loop: norm = dinv[i]^2
}

template <int F>
__global__ void k_edge_scatter(const int* __restrict__ src, const int* __restrict__ dst,
                               const float* __restrict__ hlin, const float* __restrict__ dinv,
                               float* __restrict__ out, int e) {
  long long t = (long long)blockIdx.x * blockDim.x + threadIdx.x;
  if (t >= (long long)e * F) return;
  int ei = (int)(t / F);
  int c  = (int)(t % F);
  int s = src[ei], d = dst[ei];
  atomicAdd(&out[(size_t)d * F + c], hlin[(size_t)s * F + c] * dinv[s] * dinv[d]);
}

// ---------------- orchestration ----------------

static inline unsigned nblk(long long n) { return (unsigned)((n + THREADS - 1) / THREADS); }

extern "C" void kernel_launch(void* const* d_in, const int* in_sizes, int n_in,
                              void* d_out, int out_size, void* d_ws, size_t ws_size,
                              hipStream_t stream) {
  const float* x   = (const float*)d_in[0];
  const int*   ei  = (const int*)  d_in[1];
  const float* W1  = (const float*)d_in[2];  const float* b1  = (const float*)d_in[3];
  const float* W2  = (const float*)d_in[4];  const float* b2  = (const float*)d_in[5];
  const float* W3  = (const float*)d_in[6];  const float* b3  = (const float*)d_in[7];
  const float* Wmu = (const float*)d_in[8];  const float* bmu = (const float*)d_in[9];
  const float* Wlv = (const float*)d_in[10]; const float* blv = (const float*)d_in[11];

  const int N = in_sizes[0] / 512;
  const int E = in_sizes[1] / 2;
  const int* src = ei;        // edge_index[0]
  const int* dst = ei + E;    // edge_index[1]

  // workspace carve-out
  char* ws = (char*)d_ws;
  float* dinv = (float*)ws;                 ws += ((size_t)N * 4 + 255) & ~(size_t)255;
  float* bufA = (float*)ws;                 ws += (size_t)N * 64 * 4;   // hlin1, then small bufs
  float* bufB = (float*)ws;                                            // h1 (conv1 out)
  // small buffers live inside bufA once hlin1 is consumed (needs N*28 floats < N*64)
  float* h2lin = bufA;
  float* h2    = bufA + (size_t)N * 8;
  float* h3lin = bufA + (size_t)N * 16;
  float* h3    = bufA + (size_t)N * 20;
  float* hmu   = bufA + (size_t)N * 24;
  float* hlv   = bufA + (size_t)N * 26;
  float* out_mu = (float*)d_out;
  float* out_lv = (float*)d_out + (size_t)2 * N;

  // degrees -> dinv
  k_deg_init <<<nblk(N), THREADS, 0, stream>>>(dinv, N);
  k_deg_count<<<nblk(E), THREADS, 0, stream>>>(dst, dinv, E);
  k_dinv     <<<nblk(N), THREADS, 0, stream>>>(dinv, N);

  // layer 1: WMMA GEMM, then bias+self-loop init, then edge scatter
  k_gemm_512x64_wmma<<<N / 16, 128, 0, stream>>>(x, W1, bufA);
  k_agg_init<64>    <<<nblk((long long)N * 64), THREADS, 0, stream>>>(bufA, dinv, b1, bufB, N);
  k_edge_scatter<64><<<nblk((long long)E * 64), THREADS, 0, stream>>>(src, dst, bufA, dinv, bufB, E);

  // layer 2 (relu fused into linear's input read)
  k_linear          <<<nblk((long long)N * 8), THREADS, 0, stream>>>(bufB, W2, h2lin, N, 64, 8, 1);
  k_agg_init<8>     <<<nblk((long long)N * 8), THREADS, 0, stream>>>(h2lin, dinv, b2, h2, N);
  k_edge_scatter<8> <<<nblk((long long)E * 8), THREADS, 0, stream>>>(src, dst, h2lin, dinv, h2, E);

  // layer 3
  k_linear          <<<nblk((long long)N * 4), THREADS, 0, stream>>>(h2, W3, h3lin, N, 8, 4, 1);
  k_agg_init<4>     <<<nblk((long long)N * 4), THREADS, 0, stream>>>(h3lin, dinv, b3, h3, N);
  k_edge_scatter<4> <<<nblk((long long)E * 4), THREADS, 0, stream>>>(src, dst, h3lin, dinv, h3, E);

  // mu head
  k_linear          <<<nblk((long long)N * 2), THREADS, 0, stream>>>(h3, Wmu, hmu, N, 4, 2, 1);
  k_agg_init<2>     <<<nblk((long long)N * 2), THREADS, 0, stream>>>(hmu, dinv, bmu, out_mu, N);
  k_edge_scatter<2> <<<nblk((long long)E * 2), THREADS, 0, stream>>>(src, dst, hmu, dinv, out_mu, E);

  // logvar head
  k_linear          <<<nblk((long long)N * 2), THREADS, 0, stream>>>(h3, Wlv, hlv, N, 4, 2, 1);
  k_agg_init<2>     <<<nblk((long long)N * 2), THREADS, 0, stream>>>(hlv, dinv, blv, out_lv, N);
  k_edge_scatter<2> <<<nblk((long long)E * 2), THREADS, 0, stream>>>(src, dst, hlv, dinv, out_lv, E);

  (void)n_in; (void)out_size; (void)ws_size;
}
